// ConditionalPoolingModule_62766652064078
// MI455X (gfx1250) — compile-verified, matrix operand order checked by reading, not run on hardware
//
#include <hip/hip_runtime.h>
#include <hip/hip_bf16.h>

// ---- problem dims (fixed by reference) ----
#define S_ 64
#define P_ 64
#define H_ 64
#define E_ 16
#define MID_ 512
#define BOT_ 256
#define NROWS (S_ * P_)   // 4096
#define EPS_ 1e-5f

typedef __attribute__((ext_vector_type(16))) __bf16 v16bf;
typedef __attribute__((ext_vector_type(8)))  __bf16 v8bf;
typedef __attribute__((ext_vector_type(8)))  float  v8f;
typedef __attribute__((ext_vector_type(2)))  float  v2f;

// ---------------------------------------------------------------------------
// Kernel 1: tiny prep — fold BN1 into R0'/R1', fold BN2 into bias2'
// ---------------------------------------------------------------------------
__global__ void prep_kernel(const float* __restrict__ W_emb,
                            const float* __restrict__ W1,
                            const float* __restrict__ g1, const float* __restrict__ v1,
                            const float* __restrict__ b2, const float* __restrict__ g2,
                            const float* __restrict__ be2, const float* __restrict__ m2,
                            const float* __restrict__ v2,
                            float* __restrict__ R0, float* __restrict__ R1,
                            float* __restrict__ bias2) {
    int c = threadIdx.x;
    if (c < MID_) {
        float s1 = g1[c] * rsqrtf(v1[c] + EPS_);
        float r0 = 0.f, r1 = 0.f;
        #pragma unroll
        for (int e = 0; e < E_; ++e) {
            float w1v = W1[(H_ + e) * MID_ + c];
            r0 += W_emb[0 * E_ + e] * w1v;
            r1 += W_emb[1 * E_ + e] * w1v;
        }
        R0[c] = s1 * r0;
        R1[c] = s1 * r1;
    }
    if (c < BOT_) {
        float s2 = g2[c] * rsqrtf(v2[c] + EPS_);
        bias2[c] = s2 * (b2[c] - m2[c]) + be2[c];
    }
}

// ---------------------------------------------------------------------------
// Kernel 2: swizzle W2 into per-fragment layout (BN2 column scale folded in).
// Fragment (kk, nt): 32 lanes x 16 halfs, lane L holds
//   k = kk*32 + (L/16)*8 + (t%8) + (t/8)*16 ,  n = nt*16 + (L%16)
// grid = 256 blocks (kk*16+nt), 32 threads (L)
// ---------------------------------------------------------------------------
__global__ void w2_swizzle_kernel(const float* __restrict__ W2,
                                  const float* __restrict__ g2,
                                  const float* __restrict__ v2,
                                  __bf16* __restrict__ W2sw) {
    int blk = blockIdx.x;         // kk*16 + nt
    int kk  = blk >> 4;
    int nt  = blk & 15;
    int L   = threadIdx.x;
    int n   = nt * 16 + (L & 15);
    float s2 = g2[n] * rsqrtf(v2[n] + EPS_);
    int kgrp = (L >> 4) * 8;
    __bf16* dst = W2sw + (size_t)blk * 512 + L * 16;
    #pragma unroll
    for (int t = 0; t < 16; ++t) {
        int k = kk * 32 + kgrp + (t & 7) + (t >> 3) * 16;
        dst[t] = (__bf16)(W2[k * BOT_ + n] * s2);
    }
}

// ---------------------------------------------------------------------------
// Kernel 3: P1'[row, c] = bf16( s1[c]*(b1[c] + h[row]·W1[0:64,c]
//                         + (spd*W_emb[2,:]+b_emb)·W1[64:80,c]) + t1[c] )
// grid = 4096 blocks (row), 256 threads
// ---------------------------------------------------------------------------
__global__ void p1_kernel(const float* __restrict__ h_states,
                          const float* __restrict__ speed,
                          const float* __restrict__ W_emb,
                          const float* __restrict__ b_emb,
                          const float* __restrict__ W1,
                          const float* __restrict__ b1,
                          const float* __restrict__ g1,
                          const float* __restrict__ be1,
                          const float* __restrict__ m1,
                          const float* __restrict__ v1,
                          __bf16* __restrict__ P1) {
    __shared__ float hsh[H_];
    int row = blockIdx.x;
    int tid = threadIdx.x;
    if (tid < H_) hsh[tid] = h_states[row * H_ + tid];
    __syncthreads();

    float spd = speed[row];
    float ec[E_];
    #pragma unroll
    for (int e = 0; e < E_; ++e) ec[e] = spd * W_emb[2 * E_ + e] + b_emb[e];

    #pragma unroll
    for (int half = 0; half < 2; ++half) {
        int c = tid + half * 256;
        float acc = b1[c];
        #pragma unroll 8
        for (int k = 0; k < H_; ++k) acc += hsh[k] * W1[k * MID_ + c];
        #pragma unroll
        for (int e = 0; e < E_; ++e) acc += ec[e] * W1[(H_ + e) * MID_ + c];
        float s1 = g1[c] * rsqrtf(v1[c] + EPS_);
        float t1 = be1[c] - m1[c] * s1;
        P1[(size_t)row * MID_ + c] = (__bf16)(s1 * acc + t1);
    }
}

// ---------------------------------------------------------------------------
// Kernel 4: fused main. One block per ANCHOR PAIR (s, i0=2u, i1=2u+1).
// 256 threads = 8 waves.
// Stage 1: one P1'/R0/R1 load produces BOTH anchors' relu'd bf16 rows -> LDS,
//          written as 2-wide float vectors to pull in V_PK_FMA_F32.
// Stage 2: wave w owns 8 M-tiles (2 anchors x 4) x 2 N-tiles (ntg=2w,2w+1):
//          each B fragment (global, L2-resident) feeds 8 WMMAs.
// Epilogue: per-anchor max over 4 M-tiles x 8 rows, lane-pair shfl_xor(16),
//           bias+relu, plain LDS store, two coalesced output rows.
// ---------------------------------------------------------------------------
#define KCHUNK 128
#define LPAD   8
#define LSTRIDE (KCHUNK + LPAD)   // 136 halfs per row

__global__ void __launch_bounds__(256, 2)
pool_main_kernel(const float* __restrict__ last_pos,
                 const __bf16* __restrict__ P1,
                 const __bf16* __restrict__ W2sw,
                 const float* __restrict__ R0,
                 const float* __restrict__ R1,
                 const float* __restrict__ bias2,
                 float* __restrict__ out) {
    __shared__ __bf16 A1c[2 * P_ * LSTRIDE];   // 34,816 B
    __shared__ float  smax[2][BOT_];           // 2 KB

    const int row0 = blockIdx.x * 2;           // s*64 + i0, i0 even
    const int row1 = row0 + 1;
    const int s    = row0 >> 6;
    const int i0   = row0 & 63;
    const int tid  = threadIdx.x;
    const int lane = tid & 31;
    const int wave = tid >> 5;

    // stage-1 per-thread geometry: one row j, 32 contiguous channels/chunk
    const int j1   = tid >> 2;                 // 0..63
    const int part = tid & 3;                  // 0..3 -> channels part*32..
    const float p0x = last_pos[(s * P_ + i0) * 2 + 0];
    const float p0y = last_pos[(s * P_ + i0) * 2 + 1];
    const float p1x = last_pos[(s * P_ + i0 + 1) * 2 + 0];
    const float p1y = last_pos[(s * P_ + i0 + 1) * 2 + 1];
    const float pjx = last_pos[(s * P_ + j1) * 2 + 0];
    const float pjy = last_pos[(s * P_ + j1) * 2 + 1];
    const float r0x = pjx - p0x, r0y = pjy - p0y;   // anchor 0
    const float r1x = pjx - p1x, r1y = pjy - p1y;   // anchor 1
    const __bf16* p1row = P1 + (size_t)(s * P_ + j1) * MID_;

    // stage-2 per-wave geometry: all 8 M-tiles, 2 N-tiles
    const int nt0  = wave * 2;                 // global n-tiles nt0, nt0+1
    const int mrow = lane & 15;
    const int kgrp = (lane >> 4) * 8;

    v8f acc[16];                               // [m(0..7)][nn] -> acc[m*2+nn]
    #pragma unroll
    for (int t = 0; t < 16; ++t) acc[t] = (v8f)(0.f);

    for (int chunk = 0; chunk < MID_ / KCHUNK; ++chunk) {
        // ---- stage 1: build bf16 activation chunk for BOTH anchors ----
        const int cseg  = part * 32;                 // within chunk
        const int cbase = chunk * KCHUNK + cseg;     // global channel base
        #pragma unroll
        for (int qv = 0; qv < 4; ++qv) {
            v8bf pv = *(const v8bf*)(p1row + cbase + qv * 8);
            v8bf o0, o1;
            #pragma unroll
            for (int t2 = 0; t2 < 4; ++t2) {         // 2 channels per step
                const int c = cbase + qv * 8 + t2 * 2;
                v2f base = { (float)pv[t2 * 2], (float)pv[t2 * 2 + 1] };
                v2f rc0 = *(const v2f*)(R0 + c);
                v2f rc1 = *(const v2f*)(R1 + c);
                v2f a0 = base + r0x * rc0 + r0y * rc1;   // -> v_pk_fma_f32
                v2f a1 = base + r1x * rc0 + r1y * rc1;
                o0[t2 * 2]     = (__bf16)fmaxf(a0.x, 0.f);
                o0[t2 * 2 + 1] = (__bf16)fmaxf(a0.y, 0.f);
                o1[t2 * 2]     = (__bf16)fmaxf(a1.x, 0.f);
                o1[t2 * 2 + 1] = (__bf16)fmaxf(a1.y, 0.f);
            }
            const int loff = j1 * LSTRIDE + cseg + qv * 8;
            *(v8bf*)&A1c[loff] = o0;
            *(v8bf*)&A1c[P_ * LSTRIDE + loff] = o1;
        }
        __syncthreads();

        // ---- stage 2: WMMA over this K chunk ----
        #pragma unroll
        for (int kk = 0; kk < KCHUNK / 32; ++kk) {
            const int kkg = chunk * (KCHUNK / 32) + kk;   // global k-tile
            // 2 B fragments, each reused by 8 WMMAs
            v16bf bfrag[2];
            #pragma unroll
            for (int nn = 0; nn < 2; ++nn)
                bfrag[nn] = *(const v16bf*)(W2sw +
                    (size_t)(kkg * 16 + nt0 + nn) * 512 + lane * 16);
            #pragma unroll
            for (int m = 0; m < 8; ++m) {     // m>=4 -> anchor 1
                const __bf16* arow = &A1c[(m * 16 + mrow) * LSTRIDE
                                          + kk * 32 + kgrp];
                v8bf a_lo = *(const v8bf*)(arow);
                v8bf a_hi = *(const v8bf*)(arow + 16);
                v16bf afrag = __builtin_shufflevector(a_lo, a_hi,
                    0, 1, 2, 3, 4, 5, 6, 7, 8, 9, 10, 11, 12, 13, 14, 15);
                #pragma unroll
                for (int nn = 0; nn < 2; ++nn)
                    acc[m * 2 + nn] = __builtin_amdgcn_wmma_f32_16x16x32_bf16(
                        false, afrag, false, bfrag[nn], (short)0,
                        acc[m * 2 + nn], false, false);
            }
        }
        __syncthreads();   // protect LDS before next chunk overwrites
    }

    // ---- epilogue: per-anchor max over rows, bias, relu ----
    #pragma unroll
    for (int nn = 0; nn < 2; ++nn) {
        const int n = (nt0 + nn) * 16 + (lane & 15);
        float m0 = acc[nn][0];
        float m1 = acc[8 + nn][0];
        #pragma unroll
        for (int mt = 0; mt < 4; ++mt)
            #pragma unroll
            for (int r = 0; r < 8; ++r) {
                m0 = fmaxf(m0, acc[mt * 2 + nn][r]);
                m1 = fmaxf(m1, acc[(mt + 4) * 2 + nn][r]);
            }
        // lanes L and L+16 hold same column, complementary row halves
        m0 = fmaxf(m0, __shfl_xor(m0, 16, 32));
        m1 = fmaxf(m1, __shfl_xor(m1, 16, 32));
        if (lane < 16) {
            const float bb = bias2[n];
            smax[0][n] = fmaxf(m0 + bb, 0.f);
            smax[1][n] = fmaxf(m1 + bb, 0.f);
        }
    }
    __syncthreads();

    out[(size_t)row0 * BOT_ + tid] = smax[0][tid];
    out[(size_t)row1 * BOT_ + tid] = smax[1][tid];
}

// ---------------------------------------------------------------------------
// launcher
// ---------------------------------------------------------------------------
extern "C" void kernel_launch(void* const* d_in, const int* in_sizes, int n_in,
                              void* d_out, int out_size, void* d_ws, size_t ws_size,
                              hipStream_t stream) {
    const float* h_states = (const float*)d_in[0];
    const float* last_pos = (const float*)d_in[1];
    const float* speed    = (const float*)d_in[2];
    const float* W_emb    = (const float*)d_in[3];
    const float* b_emb    = (const float*)d_in[4];
    const float* W1       = (const float*)d_in[5];
    const float* b1       = (const float*)d_in[6];
    const float* g1       = (const float*)d_in[7];
    const float* be1      = (const float*)d_in[8];
    const float* m1       = (const float*)d_in[9];
    const float* v1       = (const float*)d_in[10];
    const float* W2       = (const float*)d_in[11];
    const float* b2       = (const float*)d_in[12];
    const float* g2       = (const float*)d_in[13];
    const float* be2      = (const float*)d_in[14];
    const float* m2       = (const float*)d_in[15];
    const float* v2       = (const float*)d_in[16];
    float* out = (float*)d_out;

    // workspace carve-up
    char* wsb = (char*)d_ws;
    __bf16* P1    = (__bf16*)(wsb);                       // 4096*512*2 = 4 MB
    __bf16* W2sw  = (__bf16*)(wsb + 4194304);             // 512*256*2 = 256 KB
    float*  R0    = (float*)(wsb + 4194304 + 262144);     // 2 KB
    float*  R1    = R0 + MID_;                            // 2 KB
    float*  bias2 = R1 + MID_;                            // 1 KB

    prep_kernel<<<1, 512, 0, stream>>>(W_emb, W1, g1, v1, b2, g2, be2, m2, v2,
                                       R0, R1, bias2);
    w2_swizzle_kernel<<<256, 32, 0, stream>>>(W2, g2, v2, W2sw);
    p1_kernel<<<NROWS, 256, 0, stream>>>(h_states, speed, W_emb, b_emb,
                                         W1, b1, g1, be1, m1, v1, P1);
    pool_main_kernel<<<NROWS / 2, 256, 0, stream>>>(last_pos, P1, W2sw,
                                                    R0, R1, bias2, out);
}
